// EdgeAwareModule_68281390072577
// MI455X (gfx1250) — matrix operand order repaired and checked by reading, not compiled, not run on hardware
//
#include <hip/hip_runtime.h>
#include <hip/hip_bf16.h>
#include <math.h>

#define BB 4
#define NN 4096
#define CC 128
#define KK 16
#define PTOT (BB * NN)   // 16384
#define LN_EPS 1e-5f

typedef _Float16 v16h __attribute__((ext_vector_type(16)));
typedef _Float16 v8h  __attribute__((ext_vector_type(8)));
typedef float    v8f  __attribute__((ext_vector_type(8)));

// ---------------------------------------------------------------------------
// 3x3 symmetric eigen helpers (analytic; eigh ascending: want min & max vecs)
// ---------------------------------------------------------------------------
__device__ __forceinline__ void cross3(float ax, float ay, float az,
                                       float bx, float by, float bz,
                                       float& cx, float& cy, float& cz) {
    cx = ay * bz - az * by;
    cy = az * bx - ax * bz;
    cz = ax * by - ay * bx;
}

__device__ __forceinline__ void eigvals3(float a00, float a01, float a02,
                                         float a11, float a12, float a22,
                                         float& lmin, float& lmax) {
    float p1 = a01 * a01 + a02 * a02 + a12 * a12;
    float q  = (a00 + a11 + a22) * (1.0f / 3.0f);
    float d0 = a00 - q, d1 = a11 - q, d2 = a22 - q;
    float p2 = d0 * d0 + d1 * d1 + d2 * d2 + 2.0f * p1;
    float p  = sqrtf(fmaxf(p2 * (1.0f / 6.0f), 1e-30f));
    float iv = 1.0f / p;
    float b00 = d0 * iv, b01 = a01 * iv, b02 = a02 * iv;
    float b11 = d1 * iv, b12 = a12 * iv, b22 = d2 * iv;
    float det = b00 * (b11 * b22 - b12 * b12)
              - b01 * (b01 * b22 - b12 * b02)
              + b02 * (b01 * b12 - b11 * b02);
    float r   = fminf(fmaxf(det * 0.5f, -1.0f), 1.0f);
    float phi = acosf(r) * (1.0f / 3.0f);
    lmax = q + 2.0f * p * __cosf(phi);
    lmin = q + 2.0f * p * __cosf(phi + 2.0943951023931953f);  // +2pi/3
}

__device__ __forceinline__ void eigvec3(float a00, float a01, float a02,
                                        float a11, float a12, float a22,
                                        float lam,
                                        float& vx, float& vy, float& vz) {
    float r0x = a00 - lam, r0y = a01, r0z = a02;
    float r1x = a01, r1y = a11 - lam, r1z = a12;
    float r2x = a02, r2y = a12, r2z = a22 - lam;
    float c0x, c0y, c0z, c1x, c1y, c1z, c2x, c2y, c2z;
    cross3(r0x, r0y, r0z, r1x, r1y, r1z, c0x, c0y, c0z);
    cross3(r0x, r0y, r0z, r2x, r2y, r2z, c1x, c1y, c1z);
    cross3(r1x, r1y, r1z, r2x, r2y, r2z, c2x, c2y, c2z);
    float n0 = c0x * c0x + c0y * c0y + c0z * c0z;
    float n1 = c1x * c1x + c1y * c1y + c1z * c1z;
    float n2 = c2x * c2x + c2y * c2y + c2z * c2z;
    float bx = c0x, by = c0y, bz = c0z, bn = n0;
    if (n1 > bn) { bx = c1x; by = c1y; bz = c1z; bn = n1; }
    if (n2 > bn) { bx = c2x; by = c2y; bz = c2z; bn = n2; }
    if (bn < 1e-30f) { vx = 1.0f; vy = 0.0f; vz = 0.0f; return; }
    float inv = rsqrtf(bn);
    vx = bx * inv; vy = by * inv; vz = bz * inv;
}

// ---------------------------------------------------------------------------
// Kernel 1: KNN (top-16 by squared distance, self included) + covariance
// eigen features.  LDS-stages entire batch xyz (48 KB); one query per thread.
// ---------------------------------------------------------------------------
__global__ void __launch_bounds__(128)
knn_geo_kernel(const float* __restrict__ xyz,
               int* __restrict__ idx_out,
               float* __restrict__ gf_out) {
    __shared__ float sx[NN * 3];   // 48 KB
    const int b   = blockIdx.y;
    const int tid = threadIdx.x;
    const float* xb = xyz + (size_t)b * NN * 3;
    for (int i = tid; i < NN * 3; i += 128) sx[i] = xb[i];
    __syncthreads();

    const int q = blockIdx.x * 128 + tid;
    const float qx = sx[q * 3 + 0], qy = sx[q * 3 + 1], qz = sx[q * 3 + 2];

    float bd[KK];
    int   bi[KK];
#pragma unroll
    for (int s = 0; s < KK; ++s) { bd[s] = 3.0e38f; bi[s] = 0; }

    for (int j = 0; j < NN; ++j) {
        float dx = sx[j * 3 + 0] - qx;
        float dy = sx[j * 3 + 1] - qy;
        float dz = sx[j * 3 + 2] - qz;
        float d  = dx * dx + dy * dy + dz * dz;
        if (d < bd[KK - 1]) {             // strict <  ==> ties keep lower index
            float cd = d; int ci = j;
#pragma unroll
            for (int s = 0; s < KK; ++s) {
                if (cd < bd[s]) {
                    float td = bd[s]; int ti = bi[s];
                    bd[s] = cd; bi[s] = ci;
                    cd = td; ci = ti;
                }
            }
        }
    }

    // rel_pos statistics over the 16 neighbors (uncentered covariance / K)
    float c00 = 0, c01 = 0, c02 = 0, c11 = 0, c12 = 0, c22 = 0;
    float mx = 0, my = 0, mz = 0, md = 0;
    const size_t pbase = ((size_t)b * NN + q);
#pragma unroll
    for (int k = 0; k < KK; ++k) {
        int j = bi[k];
        float dx = sx[j * 3 + 0] - qx;
        float dy = sx[j * 3 + 1] - qy;
        float dz = sx[j * 3 + 2] - qz;
        c00 += dx * dx; c01 += dx * dy; c02 += dx * dz;
        c11 += dy * dy; c12 += dy * dz; c22 += dz * dz;
        mx += dx; my += dy; mz += dz;
        md += sqrtf(dx * dx + dy * dy + dz * dz);
        idx_out[pbase * KK + k] = j;
    }
    const float ik = 1.0f / (float)KK;
    c00 *= ik; c01 *= ik; c02 *= ik; c11 *= ik; c12 *= ik; c22 *= ik;
    mx *= ik; my *= ik; mz *= ik; md *= ik;

    float lmin, lmax;
    eigvals3(c00, c01, c02, c11, c12, c22, lmin, lmax);
    float nx, ny, nz, cx, cy, cz;
    eigvec3(c00, c01, c02, c11, c12, c22, lmin, nx, ny, nz);   // normal
    eigvec3(c00, c01, c02, c11, c12, c22, lmax, cx, cy, cz);   // curvature dir

    float* g = gf_out + pbase * 10;
    g[0] = nx; g[1] = ny; g[2] = nz;
    g[3] = cx; g[4] = cy; g[5] = cz;
    g[6] = mx; g[7] = my; g[8] = mz;
    g[9] = md;
}

// ---------------------------------------------------------------------------
// Kernel 2: mean |f - f_neighbor| over K (gather-heavy; coalesced over C)
// ---------------------------------------------------------------------------
__global__ void __launch_bounds__(128)
feat_diff_kernel(const float* __restrict__ feat,
                 const int* __restrict__ idx,
                 float* __restrict__ fd) {
    const int p = blockIdx.x;            // 0..16383
    const int b = p >> 12;               // /4096
    const int c = threadIdx.x;
    const float* fb = feat + (size_t)b * NN * CC;
    const float  f  = feat[(size_t)p * CC + c];
    float s = 0.0f;
#pragma unroll
    for (int k = 0; k < KK; ++k) {
        int j = idx[(size_t)p * KK + k];
        s += fabsf(f - fb[(size_t)j * CC + c]);
    }
    fd[(size_t)p * CC + c] = s * (1.0f / (float)KK);
}

// ---------------------------------------------------------------------------
// Kernel 3: small per-point MLP  in_dim -> 64 (LN, relu) -> 64
// Writes into edge_feat[:, col_off:col_off+64].
// ---------------------------------------------------------------------------
__global__ void __launch_bounds__(64)
mlp_ln_kernel(const float* __restrict__ in, int in_dim,
              const float* __restrict__ w1, const float* __restrict__ b1,
              const float* __restrict__ g,  const float* __restrict__ be,
              const float* __restrict__ w2, const float* __restrict__ b2,
              float* __restrict__ edge_feat, int col_off) {
    const int p = blockIdx.x;
    const int j = threadIdx.x;           // 0..63
    __shared__ float ibuf[128];
    __shared__ float hbuf[64];

    for (int i = j; i < in_dim; i += 64) ibuf[i] = in[(size_t)p * in_dim + i];
    __syncthreads();

    float h = b1[j];
    for (int i = 0; i < in_dim; ++i) h += ibuf[i] * w1[i * 64 + j];
    hbuf[j] = h;
    __syncthreads();

    float s = 0.0f, s2 = 0.0f;
#pragma unroll 8
    for (int i = 0; i < 64; ++i) { float v = hbuf[i]; s += v; s2 += v * v; }
    float m   = s * (1.0f / 64.0f);
    float var = s2 * (1.0f / 64.0f) - m * m;
    float r   = (h - m) * rsqrtf(var + LN_EPS) * g[j] + be[j];
    r = fmaxf(r, 0.0f);
    __syncthreads();
    hbuf[j] = r;
    __syncthreads();

    float o = b2[j];
#pragma unroll 8
    for (int i = 0; i < 64; ++i) o += hbuf[i] * w2[i * 64 + j];
    edge_feat[(size_t)p * 128 + col_off + j] = o;
}

// ---------------------------------------------------------------------------
// Kernel 4: edge probability  128 -> 32 (relu) -> 1 (sigmoid); wave32 reduce
// ---------------------------------------------------------------------------
__global__ void __launch_bounds__(32)
edge_prob_kernel(const float* __restrict__ ef,
                 const float* __restrict__ w1, const float* __restrict__ b1,
                 const float* __restrict__ w2, const float* __restrict__ b2,
                 float* __restrict__ prob_ws, float* __restrict__ prob_out) {
    const int p = blockIdx.x;
    const int t = threadIdx.x;           // 0..31 == one wave32
    __shared__ float sbuf[128];
    for (int i = t; i < 128; i += 32) sbuf[i] = ef[(size_t)p * 128 + i];
    __syncthreads();

    float h = b1[t];
#pragma unroll 8
    for (int i = 0; i < 128; ++i) h += sbuf[i] * w1[i * 32 + t];
    h = fmaxf(h, 0.0f);

    float v = h * w2[t];
#pragma unroll
    for (int off = 16; off > 0; off >>= 1) v += __shfl_xor(v, off);

    if (t == 0) {
        float pr = 1.0f / (1.0f + expf(-(v + b2[0])));
        prob_ws[p]  = pr;
        prob_out[p] = pr;
    }
}

// ---------------------------------------------------------------------------
// Kernel 5: refine GEMM on the CDNA5 tensor path.
//   enhanced(16384x256) @ ref_w1(256x128) + b1 -> LN -> relu -> +features
// Block: 256 threads (8 waves), 64 rows x 128 cols of output.
// K chunked by 64 through LDS (A f16 8KB, B^T f16 16KB, Y f32 32KB = 56KB).
// Each wave: row-tile (wave>>1)*16, column half (wave&1)*64, 4 col-tiles,
// accumulated with v_wmma_f32_16x16x32_f16 (f32 accumulate).
// ---------------------------------------------------------------------------
#define KCH 64

__global__ void __launch_bounds__(256)
refine_wmma_kernel(const float* __restrict__ feat,
                   const float* __restrict__ ef,
                   const float* __restrict__ prob,
                   const float* __restrict__ w1,
                   const float* __restrict__ b1,
                   const float* __restrict__ g1,
                   const float* __restrict__ be1,
                   float* __restrict__ out) {
    __shared__ __align__(16) _Float16 Alds[64 * KCH];    // [m][k]
    __shared__ __align__(16) _Float16 Blds[128 * KCH];   // [n][k] (transposed)
    __shared__ __align__(16) float    Ylds[64 * 128];

    const int tid  = threadIdx.x;
    const int wave = tid >> 5;
    const int lane = tid & 31;
    const int rowbase = blockIdx.x * 64;
    const int rt   = wave >> 1;          // row tile 0..3
    const int ch   = wave & 1;           // column half 0..1
    const int half = lane >> 4;          // K-half selector within fragment
    const int ml   = lane & 15;          // M (or N) within tile

    v8f acc[4];
#pragma unroll
    for (int t = 0; t < 4; ++t)
#pragma unroll
        for (int e = 0; e < 8; ++e) acc[t][e] = 0.0f;

    for (int k0 = 0; k0 < 256; k0 += KCH) {
        __syncthreads();
        // Stage A chunk: enhanced = [features | edge_feat * prob], cast f16
        for (int x = tid; x < 64 * KCH; x += 256) {
            int m = x >> 6, k = x & (KCH - 1);
            int row = rowbase + m, kg = k0 + k;
            float v = (kg < 128)
                        ? feat[(size_t)row * 128 + kg]
                        : ef[(size_t)row * 128 + (kg - 128)] * prob[row];
            Alds[m * KCH + k] = (_Float16)v;
        }
        // Stage B chunk transposed: Blds[n][k] = w1[k0+k][n], cast f16
        for (int x = tid; x < 128 * KCH; x += 256) {
            int k = x >> 7, n = x & 127;
            Blds[n * KCH + k] = (_Float16)w1[(size_t)(k0 + k) * 128 + n];
        }
        __syncthreads();

#pragma unroll
        for (int ks = 0; ks < 2; ++ks) {             // two 32-wide K steps
            // A fragment: lane holds row rt*16+ml, K = ks*32 + half*8 + j (+16)
            const _Float16* ap = &Alds[(rt * 16 + ml) * KCH + ks * 32 + half * 8];
            v16h av;
            *((v8h*)&av)     = *((const v8h*)ap);
            *((v8h*)&av + 1) = *((const v8h*)(ap + 16));
#pragma unroll
            for (int ct = 0; ct < 4; ++ct) {
                // B fragment: lane holds col n, K = ks*32 + half*8 + j (+16)
                int n = ch * 64 + ct * 16 + ml;
                const _Float16* bp = &Blds[n * KCH + ks * 32 + half * 8];
                v16h bv;
                *((v8h*)&bv)     = *((const v8h*)bp);
                *((v8h*)&bv + 1) = *((const v8h*)(bp + 16));
                acc[ct] = __builtin_amdgcn_wmma_f32_16x16x32_f16(
                    false, av, false, bv, (short)0, acc[ct], false, false);
            }
        }
    }

    // Spill accumulators (+bias) to LDS for the row-wise LayerNorm.
    __syncthreads();
#pragma unroll
    for (int ct = 0; ct < 4; ++ct) {
        int n = ch * 64 + ct * 16 + ml;
#pragma unroll
        for (int r = 0; r < 8; ++r) {
            int m = rt * 16 + r + half * 8;          // C/D layout (ISA 7.12.2)
            Ylds[m * 128 + n] = acc[ct][r] + b1[n];
        }
    }
    __syncthreads();

    // LayerNorm + relu + residual: 4 lanes per row, wave32 shuffle combine.
    {
        const int row  = tid >> 2;                   // 0..63
        const int part = tid & 3;
        float s = 0.0f, s2 = 0.0f;
#pragma unroll
        for (int i = 0; i < 32; ++i) {
            float v = Ylds[row * 128 + part * 32 + i];
            s += v; s2 += v * v;
        }
        s  += __shfl_xor(s, 1);  s  += __shfl_xor(s, 2);
        s2 += __shfl_xor(s2, 1); s2 += __shfl_xor(s2, 2);
        float m   = s * (1.0f / 128.0f);
        float var = s2 * (1.0f / 128.0f) - m * m;
        float inv = rsqrtf(var + LN_EPS);
        const size_t grow = (size_t)(rowbase + row) * 128;
#pragma unroll
        for (int i = 0; i < 32; ++i) {
            int c = part * 32 + i;
            float v = (Ylds[row * 128 + c] - m) * inv * g1[c] + be1[c];
            out[grow + c] = fmaxf(v, 0.0f) + feat[grow + c];
        }
    }
}

// ---------------------------------------------------------------------------
extern "C" void kernel_launch(void* const* d_in, const int* in_sizes, int n_in,
                              void* d_out, int out_size, void* d_ws, size_t ws_size,
                              hipStream_t stream) {
    const float* xyz      = (const float*)d_in[0];
    const float* features = (const float*)d_in[1];
    const float* geo_w1   = (const float*)d_in[2];
    const float* geo_b1   = (const float*)d_in[3];
    const float* geo_g1   = (const float*)d_in[4];
    const float* geo_be1  = (const float*)d_in[5];
    const float* geo_w2   = (const float*)d_in[6];
    const float* geo_b2   = (const float*)d_in[7];
    const float* diff_w1  = (const float*)d_in[8];
    const float* diff_b1  = (const float*)d_in[9];
    const float* diff_g1  = (const float*)d_in[10];
    const float* diff_be1 = (const float*)d_in[11];
    const float* diff_w2  = (const float*)d_in[12];
    const float* diff_b2  = (const float*)d_in[13];
    const float* ep_w1    = (const float*)d_in[14];
    const float* ep_b1    = (const float*)d_in[15];
    const float* ep_w2    = (const float*)d_in[16];
    const float* ep_b2    = (const float*)d_in[17];
    const float* ref_w1   = (const float*)d_in[18];
    const float* ref_b1   = (const float*)d_in[19];
    const float* ref_g1   = (const float*)d_in[20];
    const float* ref_be1  = (const float*)d_in[21];

    float* out = (float*)d_out;                       // refined | edge_prob
    const size_t P = (size_t)PTOT;

    // Workspace layout (~18.5 MB)
    int*   ws_idx = (int*)d_ws;                       // P*16 ints
    float* ws_gf  = (float*)((char*)d_ws + P * KK * sizeof(int));  // P*10
    float* ws_fd  = ws_gf + P * 10;                   // P*128
    float* ws_ef  = ws_fd + P * 128;                  // P*128
    float* ws_pr  = ws_ef + P * 128;                  // P

    knn_geo_kernel<<<dim3(NN / 128, BB), 128, 0, stream>>>(xyz, ws_idx, ws_gf);
    feat_diff_kernel<<<(int)P, 128, 0, stream>>>(features, ws_idx, ws_fd);
    mlp_ln_kernel<<<(int)P, 64, 0, stream>>>(ws_gf, 10,
        geo_w1, geo_b1, geo_g1, geo_be1, geo_w2, geo_b2, ws_ef, 0);
    mlp_ln_kernel<<<(int)P, 64, 0, stream>>>(ws_fd, 128,
        diff_w1, diff_b1, diff_g1, diff_be1, diff_w2, diff_b2, ws_ef, 64);
    edge_prob_kernel<<<(int)P, 32, 0, stream>>>(ws_ef,
        ep_w1, ep_b1, ep_w2, ep_b2, ws_pr, out + P * 128);
    refine_wmma_kernel<<<(int)(P / 64), 256, 0, stream>>>(features, ws_ef, ws_pr,
        ref_w1, ref_b1, ref_g1, ref_be1, out);
}
